// Vision_Transformer_36541581754469
// MI455X (gfx1250) — compile-verified, matrix operand order checked by reading, not run on hardware
//
#include <hip/hip_runtime.h>

// ---------------- types & helpers ----------------
typedef __attribute__((ext_vector_type(16))) __bf16 v16bf;
typedef __attribute__((ext_vector_type(8)))  __bf16 v8bf;
typedef __attribute__((ext_vector_type(8)))  float  v8f;
typedef __attribute__((ext_vector_type(4)))  int    v4i;

// Native conversion: lowers to v_cvt_pk_bf16_f32 on gfx1250 (RTNE).
__device__ __forceinline__ __bf16 f2bf(float f) { return (__bf16)f; }

// CDNA5 async global->LDS copy (ASYNCcnt-tracked), with portable fallback.
#if __has_builtin(__builtin_amdgcn_global_load_async_to_lds_b128) && \
    __has_builtin(__builtin_amdgcn_s_wait_asynccnt)
#define ATTN_ASYNC 1
#endif

__device__ __forceinline__ void stage16B(__bf16* lds, const __bf16* glob) {
#ifdef ATTN_ASYNC
  __builtin_amdgcn_global_load_async_to_lds_b128(
      (__attribute__((address_space(1))) v4i*)glob,
      (__attribute__((address_space(3))) v4i*)lds, 0, 0);
#else
  *(v8bf*)lds = *(const v8bf*)glob;
#endif
}
__device__ __forceinline__ void stage_fence() {
#ifdef ATTN_ASYNC
  __builtin_amdgcn_s_wait_asynccnt(0);
#endif
}

// Load a 16x32 bf16 WMMA A/B fragment from an LDS tile.
// 16-bit A layout (ISA 7.12.2): lane l -> row = base + (l&15); lane-half g = l>>4
// needs K-chunks [cbase+8g, +8) in elems 0..7 and [cbase+16+8g, +8) in 8..15.
template <int PITCH>
__device__ __forceinline__ v16bf ld_frag(const __bf16 (*tile)[PITCH], int row,
                                         int cbase, int lane) {
  const __bf16* p0 = &tile[row][cbase + ((lane >> 4) << 3)];
  const __bf16* p1 = p0 + 16;
  v16bf a;
#pragma unroll
  for (int i = 0; i < 8; i++) { a[i] = p0[i]; a[8 + i] = p1[i]; }
  return a;
}

// ---------------- generic WMMA GEMM: C = A * W^T (+bias)(+res)(+gelu) --------
// A: [M,K] lda, W: [N,K] ldw, C/Cbf: [M,N] ldc (exactly one non-null).
// 256 threads = 8 waves; block tile 128x128; wave owns 32x64 (acc 2x4).
// Double-buffered LDS + register prefetch: one barrier per k-step.
__global__ __launch_bounds__(256) void gemm_bf16_wmma(
    const float* __restrict__ A, int lda,
    const float* __restrict__ W, int ldw,
    float* __restrict__ C, __bf16* __restrict__ Cbf, int ldc,
    const float* __restrict__ bias,
    const float* __restrict__ Res, int ldr,
    int M, int N, int K, int do_gelu) {
  __shared__ __bf16 As[2][128][40];
  __shared__ __bf16 Ws[2][128][40];
  const int t = threadIdx.x;
  const int lane = t & 31, wave = t >> 5;
  const int wm = (wave >> 1) * 32;   // 4 wave-rows
  const int wn = (wave & 1) * 64;    // 2 wave-cols
  const int half = lane >> 4, nl = lane & 15;
  const int m0 = blockIdx.y * 128, n0 = blockIdx.x * 128;
  const int lrow = t >> 1, lcol = (t & 1) * 16;

  const bool aval = (m0 + lrow) < M;
  const bool wval = (n0 + lrow) < N;
  const float* aptr = A + (size_t)(m0 + lrow) * lda + lcol;
  const float* wptr = W + (size_t)(n0 + lrow) * ldw + lcol;

  float ar[16], wr[16];
  const float4 fz = {0.f, 0.f, 0.f, 0.f};

  auto fetch = [&](int k0) {
#pragma unroll
    for (int i = 0; i < 4; i++) {
      float4 fa = aval ? *(const float4*)(aptr + k0 + 4 * i) : fz;
      ar[4 * i + 0] = fa.x; ar[4 * i + 1] = fa.y; ar[4 * i + 2] = fa.z; ar[4 * i + 3] = fa.w;
      float4 fw = wval ? *(const float4*)(wptr + k0 + 4 * i) : fz;
      wr[4 * i + 0] = fw.x; wr[4 * i + 1] = fw.y; wr[4 * i + 2] = fw.z; wr[4 * i + 3] = fw.w;
    }
  };
  auto stash = [&](int buf) {
    v8bf a0, a1, w0, w1;
#pragma unroll
    for (int i = 0; i < 8; i++) {
      a0[i] = f2bf(ar[i]); a1[i] = f2bf(ar[8 + i]);
      w0[i] = f2bf(wr[i]); w1[i] = f2bf(wr[8 + i]);
    }
    *(v8bf*)&As[buf][lrow][lcol]     = a0;
    *(v8bf*)&As[buf][lrow][lcol + 8] = a1;
    *(v8bf*)&Ws[buf][lrow][lcol]     = w0;
    *(v8bf*)&Ws[buf][lrow][lcol + 8] = w1;
  };

  v8f acc[2][4] = {};
  const int nk = K >> 5;

  fetch(0);
  stash(0);
  __syncthreads();

  for (int kt = 0; kt < nk; kt++) {
    const int buf = kt & 1;
    if (kt + 1 < nk) fetch((kt + 1) << 5);   // prefetch next tile

    v16bf af[2];
#pragma unroll
    for (int i = 0; i < 2; i++) af[i] = ld_frag<40>(As[buf], wm + 16 * i + nl, 0, lane);
#pragma unroll
    for (int j = 0; j < 4; j++) {
      v16bf b = ld_frag<40>(Ws[buf], wn + 16 * j + nl, 0, lane);
#pragma unroll
      for (int i = 0; i < 2; i++)
        acc[i][j] = __builtin_amdgcn_wmma_f32_16x16x32_bf16(
            false, af[i], false, b, (short)0, acc[i][j], false, false);
    }

    if (kt + 1 < nk) stash(buf ^ 1);
    __syncthreads();
  }

  // epilogue — C layout (ISA 7.12.2): reg r -> m = 8*(l>>4)+r, n = l&15
#pragma unroll
  for (int i = 0; i < 2; i++) {
#pragma unroll
    for (int j = 0; j < 4; j++) {
      int gn = n0 + wn + 16 * j + nl;
      if (gn >= N) continue;
      float bv = bias ? bias[gn] : 0.f;
#pragma unroll
      for (int r = 0; r < 8; r++) {
        int gm = m0 + wm + 16 * i + 8 * half + r;
        if (gm >= M) continue;
        float v = acc[i][j][r] + bv;
        if (Res) v += Res[(size_t)gm * ldr + gn];
        if (do_gelu) v = 0.5f * v * (1.0f + erff(v * 0.70710678118f));
        if (Cbf) Cbf[(size_t)gm * ldc + gn] = f2bf(v);
        else     C  [(size_t)gm * ldc + gn] = v;
      }
    }
  }
}

// ---------------- flash attention (per b,h,q-tile of 64) --------------------
// q,k,v are bf16 [B,S,H*D] head-major. softmax(Q K^T / 8) V, S=577, D=64.
__global__ void attention_flash(const __bf16* __restrict__ Qb,
                                const __bf16* __restrict__ Kb,
                                const __bf16* __restrict__ Vb,
                                float* __restrict__ O) {
  const int S = 577, E = 768;
  const int bh = blockIdx.x, b = bh / 12, h = bh % 12;
  const int q0 = blockIdx.y * 64;

  __shared__ __bf16 Qs[64][72];
  __shared__ __bf16 KVs[64][72];   // K tile, then reused for V^T tile
  __shared__ float  Ss[64][68];    // raw scores for current 64x64 tile
  __shared__ __bf16 Ps[64][72];    // exp probabilities (bf16)
  __shared__ float  rowmax[64], rowsum[64], rowscale[64];
  __shared__ float  pred[64][2];   // per-half row reductions

  const int t = threadIdx.x, lane = t & 31, wave = t >> 5;
  const int wm = (wave >> 1) * 32, wn = (wave & 1) * 32;
  const int half = lane >> 4, nl = lane & 15;
  const int lrow = t >> 1, lc0 = (t & 1) * 32;
  const int r = t >> 1, hf = t & 1;   // softmax: 2 threads per row

  {  // stage Q tile (async global->LDS; rows >= S clamp to row 0, discarded)
    int s = q0 + lrow, scl = s < S ? s : 0;
    const __bf16* src = Qb + (size_t)(b * S + scl) * E + h * 64 + lc0;
#pragma unroll
    for (int i = 0; i < 4; i++) stage16B(&Qs[lrow][lc0 + 8 * i], src + 8 * i);
  }
  if (t < 64) { rowmax[t] = -1e30f; rowsum[t] = 0.f; }
  stage_fence();
  __syncthreads();

  v8f oacc[2][2] = {};

  for (int kt = 0; kt < 10; kt++) {   // ceil(577/64) = 10 key tiles
    {  // stage K tile (clamped rows are masked by column test below)
      int s = kt * 64 + lrow, scl = s < S ? s : 0;
      const __bf16* src = Kb + (size_t)(b * S + scl) * E + h * 64 + lc0;
#pragma unroll
      for (int i = 0; i < 4; i++) stage16B(&KVs[lrow][lc0 + 8 * i], src + 8 * i);
    }
    stage_fence();
    __syncthreads();

    v8f sacc[2][2] = {};
#pragma unroll
    for (int ks = 0; ks < 2; ks++) {     // D = 64 = 2 K-steps of 32
#pragma unroll
      for (int i = 0; i < 2; i++) {
        v16bf a = ld_frag<72>(Qs, wm + 16 * i + nl, ks * 32, lane);
#pragma unroll
        for (int j = 0; j < 2; j++) {
          v16bf bb = ld_frag<72>(KVs, wn + 16 * j + nl, ks * 32, lane);
          sacc[i][j] = __builtin_amdgcn_wmma_f32_16x16x32_bf16(
              false, a, false, bb, (short)0, sacc[i][j], false, false);
        }
      }
    }
    __syncthreads();   // done reading K tile; KVs will be overwritten with V^T

    // scatter scores to LDS (scaled by 1/sqrt(64)), mask OOB keys
#pragma unroll
    for (int i = 0; i < 2; i++)
#pragma unroll
      for (int j = 0; j < 2; j++)
#pragma unroll
        for (int rr = 0; rr < 8; rr++) {
          int rl = wm + 16 * i + 8 * half + rr;
          int cl = wn + 16 * j + nl;
          Ss[rl][cl] = (kt * 64 + cl < S) ? sacc[i][j][rr] * 0.125f : -1e30f;
        }
    {  // stage V^T tile into KVs: KVs[d][k_local] (bf16 gather, no cvt)
      int d = lrow;
#pragma unroll
      for (int i = 0; i < 32; i++) {
        int kk = lc0 + i, s = kt * 64 + kk;
        KVs[d][kk] = (s < S) ? Vb[(size_t)(b * S + s) * E + h * 64 + d] : (__bf16)0.f;
      }
    }
    __syncthreads();

    // online softmax: 2 threads per row, 32 cols each
    {
      float mloc = -1e30f;
      for (int c = 0; c < 32; c++) mloc = fmaxf(mloc, Ss[r][hf * 32 + c]);
      pred[r][hf] = mloc;
      __syncthreads();
      float mold = rowmax[r];
      float mnew = fmaxf(mold, fmaxf(pred[r][0], pred[r][1]));
      float sm = 0.f;
      for (int c = 0; c < 32; c++) {
        float p = __expf(Ss[r][hf * 32 + c] - mnew);
        Ps[r][hf * 32 + c] = f2bf(p);
        sm += p;
      }
      __syncthreads();
      pred[r][hf] = sm;
      __syncthreads();
      if (hf == 0) {
        float scl = __expf(mold - mnew);
        rowsum[r] = rowsum[r] * scl + pred[r][0] + pred[r][1];
        rowmax[r] = mnew;
        rowscale[r] = scl;
      }
      __syncthreads();
    }

    // rescale running output, then O += P @ V
#pragma unroll
    for (int i = 0; i < 2; i++) {
      float sc[8];
#pragma unroll
      for (int rr = 0; rr < 8; rr++) sc[rr] = rowscale[wm + 16 * i + 8 * half + rr];
#pragma unroll
      for (int j = 0; j < 2; j++)
#pragma unroll
        for (int rr = 0; rr < 8; rr++) oacc[i][j][rr] *= sc[rr];
    }
#pragma unroll
    for (int ks = 0; ks < 2; ks++) {
#pragma unroll
      for (int i = 0; i < 2; i++) {
        v16bf a = ld_frag<72>(Ps, wm + 16 * i + nl, ks * 32, lane);
#pragma unroll
        for (int j = 0; j < 2; j++) {
          v16bf bb = ld_frag<72>(KVs, wn + 16 * j + nl, ks * 32, lane);
          oacc[i][j] = __builtin_amdgcn_wmma_f32_16x16x32_bf16(
              false, a, false, bb, (short)0, oacc[i][j], false, false);
        }
      }
    }
    __syncthreads();
  }

#pragma unroll
  for (int i = 0; i < 2; i++)
#pragma unroll
    for (int j = 0; j < 2; j++)
#pragma unroll
      for (int rr = 0; rr < 8; rr++) {
        int rl = wm + 16 * i + 8 * half + rr;
        int q = q0 + rl;
        if (q < S) {
          int d = wn + 16 * j + nl;
          O[(size_t)(b * S + q) * E + h * 64 + d] = oacc[i][j][rr] / rowsum[rl];
        }
      }
}

// ---------------- layernorm (one block per row, E=768) ----------------------
__global__ void layernorm_k(const float* __restrict__ x, const float* __restrict__ sc,
                            const float* __restrict__ bs, float* __restrict__ o) {
  __shared__ float red[256];
  const int row = blockIdx.x, t = threadIdx.x;
  const float* xr = x + (size_t)row * 768;
  float a0 = xr[t], a1 = xr[t + 256], a2 = xr[t + 512];
  red[t] = a0 + a1 + a2;
  __syncthreads();
  for (int off = 128; off > 0; off >>= 1) { if (t < off) red[t] += red[t + off]; __syncthreads(); }
  float mu = red[0] * (1.f / 768.f);
  __syncthreads();
  float d0 = a0 - mu, d1 = a1 - mu, d2 = a2 - mu;
  red[t] = d0 * d0 + d1 * d1 + d2 * d2;
  __syncthreads();
  for (int off = 128; off > 0; off >>= 1) { if (t < off) red[t] += red[t + off]; __syncthreads(); }
  float inv = rsqrtf(red[0] * (1.f / 768.f) + 1e-5f);
  float* orow = o + (size_t)row * 768;
  orow[t]       = d0 * inv * sc[t]       + bs[t];
  orow[t + 256] = d1 * inv * sc[t + 256] + bs[t + 256];
  orow[t + 512] = d2 * inv * sc[t + 512] + bs[t + 512];
}

// ---------------- patch packing: [B,3,384,384] -> [B*576, 768] --------------
__global__ void pack_patches(const float* __restrict__ x, float* __restrict__ Xp) {
  int idx = blockIdx.x * 256 + threadIdx.x;
  if (idx >= 16 * 576 * 768) return;
  int col = idx % 768, row = idx / 768;
  int p = row % 576, b = row / 576;
  int c = col >> 8, rem = col & 255;
  int i = rem >> 4, j = rem & 15;
  int hp = p / 24, wp = p % 24;
  Xp[idx] = x[((size_t)(b * 3 + c) * 384 + hp * 16 + i) * 384 + wp * 16 + j];
}

// ---------------- cls token + positional embedding --------------------------
__global__ void add_cls_pos(float* __restrict__ xb, const float* __restrict__ cls,
                            const float* __restrict__ pos) {
  int idx = blockIdx.x * 256 + threadIdx.x;
  if (idx >= 16 * 577 * 768) return;
  int e = idx % 768, s = (idx / 768) % 577;
  if (s == 0) xb[idx] = cls[e] + pos[e];
  else        xb[idx] += pos[s * 768 + e];
}

// ---------------- classifier head (tiny: 16 x 1000 x 768) -------------------
__global__ void head_k(const float* __restrict__ xb, const float* __restrict__ hw,
                       const float* __restrict__ hb, float* __restrict__ out) {
  int b = blockIdx.x;
  int n = blockIdx.y * 128 + threadIdx.x;
  if (n >= 1000) return;
  const float* xr = xb + (size_t)b * 577 * 768;   // cls row (s = 0)
  const float* wr = hw + (size_t)n * 768;
  float s = hb[n];
  for (int k = 0; k < 768; k++) s += xr[k] * wr[k];
  out[b * 1000 + n] = s;
}

// ---------------- host orchestration ----------------------------------------
extern "C" void kernel_launch(void* const* d_in, const int* in_sizes, int n_in,
                              void* d_out, int out_size, void* d_ws, size_t ws_size,
                              hipStream_t stream) {
  const float* x      = (const float*)d_in[0];
  const float* conv_w = (const float*)d_in[1];
  const float* conv_b = (const float*)d_in[2];
  const float* cls    = (const float*)d_in[3];
  const float* pos    = (const float*)d_in[4];
  const float* ln1_s  = (const float*)d_in[5];
  const float* ln1_b  = (const float*)d_in[6];
  const float* wq     = (const float*)d_in[7];
  const float* wk     = (const float*)d_in[8];
  const float* wv     = (const float*)d_in[9];
  const float* wo     = (const float*)d_in[10];
  const float* bo     = (const float*)d_in[11];
  const float* ln2_s  = (const float*)d_in[12];
  const float* ln2_b  = (const float*)d_in[13];
  const float* w1     = (const float*)d_in[14];
  const float* b1     = (const float*)d_in[15];
  const float* w2     = (const float*)d_in[16];
  const float* b2     = (const float*)d_in[17];
  const float* hw     = (const float*)d_in[18];
  const float* hb     = (const float*)d_in[19];
  float* out = (float*)d_out;
  float* ws  = (float*)d_ws;

  const int Bn = 16, S = 577, E = 768, Hh = 12, D = 64, NP = 576, NC = 1000;
  const int M = Bn * S;                 // 9232 token rows
  const size_t ME = (size_t)M * E;
  float*  xbuf  = ws;                   // residual stream
  float*  hbuf  = ws + 1 * ME;          // LN output (also patch matrix at start)
  __bf16* qbuf  = (__bf16*)(ws + 2 * ME);
  __bf16* kbuf  = (__bf16*)(ws + 3 * ME);
  __bf16* vbuf  = (__bf16*)(ws + 4 * ME);
  float*  aobuf = ws + 5 * ME;
  float*  resbf = ws + 6 * ME;
  float*  midbf = ws + 7 * ME;          // [M, 3072]

  // patch embedding: pack, then per-batch GEMM against conv_w [768,768]
  {
    int tot = Bn * NP * E;
    pack_patches<<<(tot + 255) / 256, 256, 0, stream>>>(x, hbuf);
  }
  for (int b = 0; b < Bn; b++) {
    gemm_bf16_wmma<<<dim3((E + 127) / 128, (NP + 127) / 128), 256, 0, stream>>>(
        hbuf + (size_t)b * NP * E, E, conv_w, E,
        xbuf + (size_t)(b * S + 1) * E, nullptr, E, conv_b, nullptr, 0, NP, E, E, 0);
  }
  {
    int tot = M * E;
    add_cls_pos<<<(tot + 255) / 256, 256, 0, stream>>>(xbuf, cls, pos);
  }

  for (int l = 0; l < 12; l++) {
    layernorm_k<<<M, 256, 0, stream>>>(xbuf, ln1_s + l * E, ln1_b + l * E, hbuf);

    // q/k/v: shared per-head weight -> [M*H, 64] x [64,64]^T, bf16 output
    const int MH = M * Hh;
    gemm_bf16_wmma<<<dim3(1, (MH + 127) / 128), 256, 0, stream>>>(
        hbuf, D, wq + (size_t)l * D * D, D, nullptr, qbuf, D,
        nullptr, nullptr, 0, MH, D, D, 0);
    gemm_bf16_wmma<<<dim3(1, (MH + 127) / 128), 256, 0, stream>>>(
        hbuf, D, wk + (size_t)l * D * D, D, nullptr, kbuf, D,
        nullptr, nullptr, 0, MH, D, D, 0);
    gemm_bf16_wmma<<<dim3(1, (MH + 127) / 128), 256, 0, stream>>>(
        hbuf, D, wv + (size_t)l * D * D, D, nullptr, vbuf, D,
        nullptr, nullptr, 0, MH, D, D, 0);

    attention_flash<<<dim3(Bn * Hh, (S + 63) / 64), 128, 0, stream>>>(
        qbuf, kbuf, vbuf, aobuf);

    // output projection + residual
    gemm_bf16_wmma<<<dim3((E + 127) / 128, (M + 127) / 128), 256, 0, stream>>>(
        aobuf, E, wo + (size_t)l * E * E, E, resbf, nullptr, E,
        bo + l * E, xbuf, E, M, E, E, 0);

    layernorm_k<<<M, 256, 0, stream>>>(resbf, ln2_s + l * E, ln2_b + l * E, hbuf);

    // MLP up + GELU
    gemm_bf16_wmma<<<dim3(3072 / 128, (M + 127) / 128), 256, 0, stream>>>(
        hbuf, E, w1 + (size_t)l * 3072 * E, E, midbf, nullptr, 3072,
        b1 + l * 3072, nullptr, 0, M, 3072, E, 1);
    // MLP down + residual
    gemm_bf16_wmma<<<dim3((E + 127) / 128, (M + 127) / 128), 256, 0, stream>>>(
        midbf, 3072, w2 + (size_t)l * E * 3072, 3072, xbuf, nullptr, E,
        b2 + l * E, resbf, E, M, E, 3072, 0);
  }

  head_k<<<dim3(Bn, (NC + 127) / 128), 128, 0, stream>>>(xbuf, hw, hb, out);
}
